// Encoder_1623497638443
// MI455X (gfx1250) — matrix operand order repaired and checked by reading, not compiled, not run on hardware
//
#include <hip/hip_runtime.h>
#include <hip/hip_bf16.h>

// ---------------------------------------------------------------------------
// Types for CDNA5 WMMA (wave32): A/B = 16x(bf16) = 8 VGPRs, C/D = 8x(f32)
// ---------------------------------------------------------------------------
typedef __attribute__((ext_vector_type(16))) __bf16 v16bf;
typedef __attribute__((ext_vector_type(8)))  float  v8f;
typedef __attribute__((ext_vector_type(4)))  unsigned int uint4v;
typedef __attribute__((ext_vector_type(4)))  unsigned int u32x4;
typedef __attribute__((ext_vector_type(4)))  int i32x4;
typedef __attribute__((ext_vector_type(8)))  int i32x8;

union BF16Frag { uint4v u[2]; v16bf v; };

__device__ __forceinline__ float sigf(float x) {
  return 1.0f / (1.0f + __expf(-x));
}
__device__ __forceinline__ float tanh_fast(float x) {
  // tanh(x) = 1 - 2/(1+exp(2x)); saturates correctly to +/-1 on overflow
  return 1.0f - 2.0f / (1.0f + __expf(2.0f * x));
}

// Problem sizes
#define BB 64
#define FF 32
#define TT 128
#define HH 128
#define EE 100
#define EPAD 112
#define GG 512          // 4*H
#define NSTEP 4096      // T*F
#define W1LDS_BYTES (GG * 256 * 2)   // 262144: W1cat bf16 in LDS

// ---------------------------------------------------------------------------
// Prep kernels: convert weights/inputs to bf16 into workspace.
// ---------------------------------------------------------------------------
__global__ void k_cvt_bf16(const float* __restrict__ in, __bf16* __restrict__ out, int n) {
  int i = blockIdx.x * blockDim.x + threadIdx.x;
  if (i < n) out[i] = (__bf16)in[i];
}

// W1cat[n][k], k in [0,256): first 128 = W_ih1 row n, next 128 = W_hh1 row n
__global__ void k_prep_w1cat(const float* __restrict__ wih1, const float* __restrict__ whh1,
                             __bf16* __restrict__ out) {
  int i = blockIdx.x * blockDim.x + threadIdx.x;
  if (i >= GG * 256) return;
  int n = i >> 8, k = i & 255;
  float v = (k < HH) ? wih1[n * HH + k] : whh1[n * HH + (k - HH)];
  out[i] = (__bf16)v;
}

// WsT[f][e][t] (E padded to 112, zero fill) from Ws[f][t][e]
__global__ void k_prep_wst(const float* __restrict__ ws, __bf16* __restrict__ out) {
  int i = blockIdx.x * blockDim.x + threadIdx.x;
  if (i >= FF * EPAD * TT) return;
  int f = i / (EPAD * TT);
  int r = i % (EPAD * TT);
  int e = r / TT, t = r % TT;
  float v = (e < EE) ? ws[(f * TT + t) * EE + e] : 0.0f;
  out[i] = (__bf16)v;
}

// ---------------------------------------------------------------------------
// Spatial embedding: per f, (64x128)@(128x100) + bias via v_wmma bf16.
// One wave per (f, mblock, etile): 32*4*7 = 896 waves.
// ---------------------------------------------------------------------------
__global__ __launch_bounds__(256) void k_spatial(
    const __bf16* __restrict__ xbf,   // (B,F,T) bf16
    const __bf16* __restrict__ wst,   // (F,112,T) bf16
    const float*  __restrict__ bs,    // (F,E) f32
    float* __restrict__ out_sp)       // (B,F,E) f32
{
  int gw   = blockIdx.x * 8 + (threadIdx.x >> 5);
  int lane = threadIdx.x & 31;
  int ln = lane & 15, hiHalf = lane >> 4;
  int f  = gw / 28, rem = gw % 28;
  int mb = rem / 7, et = rem % 7;
  int n  = et * 16 + ln;                       // output column (e), padded ok

  const __bf16* arow = xbf + ((mb * 16 + ln) * FF + f) * TT;  // A row: batch m
  const __bf16* brow = wst + (f * EPAD + n) * TT;             // B col n, k-contig

  v8f acc = {0.f,0.f,0.f,0.f,0.f,0.f,0.f,0.f};
#pragma unroll
  for (int j = 0; j < 4; ++j) {
    int kA = 32 * j + hiHalf * 8;      // A: two 8-elem chunks at kA and kA+16
    BF16Frag a;
    const uint4v* pa = (const uint4v*)(arow + kA);
    a.u[0] = pa[0]; a.u[1] = pa[2];
    int kB = 32 * j + hiHalf * 8;      // B: two 8-elem chunks at kB and kB+16
    BF16Frag b;
    const uint4v* pb = (const uint4v*)(brow + kB);
    b.u[0] = pb[0]; b.u[1] = pb[2];
    acc = __builtin_amdgcn_wmma_f32_16x16x32_bf16(false, a.v, false, b.v,
                                                  (short)0, acc, false, false);
  }
  if (n < EE) {
    float bias = bs[f * EE + n];
#pragma unroll
    for (int i = 0; i < 8; ++i) {
      int row = mb * 16 + i + 8 * hiHalf;     // batch index
      out_sp[(row * FF + f) * EE + n] = acc[i] + bias;
    }
  }
}

// ---------------------------------------------------------------------------
// Persistent 2-layer LSTM: one workgroup (32 waves), 4096 steps.
// Wave w -> (mb = w>>3 : 16 batch rows, hb = w&7 : 16 hidden units).
// h0/h1 in LDS (bf16, A operand); c0/c1 in accumulator-layout registers.
// Layer-1 weights (256 KB bf16) are staged once into LDS by the Tensor Data
// Mover (tensor_load_to_lds + s_wait_tensorcnt); layer-0 weights stream from
// L2 (they total 128 KB and are re-read at identical addresses every step).
// ---------------------------------------------------------------------------
__global__ __launch_bounds__(1024) void k_lstm(
    const float* __restrict__ x_in,    // (B,F,T) f32
    const __bf16* __restrict__ whh0,   // (512,128) bf16, [n][k]
    const __bf16* __restrict__ w1cat,  // (512,256) bf16, [n][k] (ih1|hh1)
    const float* __restrict__ w_ih0,   // (512,1)
    const float* __restrict__ b_ih0, const float* __restrict__ b_hh0,
    const float* __restrict__ b_ih1, const float* __restrict__ b_hh1,
    float* __restrict__ out_tmp)       // (T,B,H) f32
{
  __shared__ __bf16 hcat[BB * 256];    // [b][k]: k<128 = h0, k>=128 = h1 (32 KB)
  __shared__ float  xbuf[BB];
  extern __shared__ __bf16 w1lds[];    // 512x256 bf16 = 256 KB (dynamic LDS)

  const int tid  = threadIdx.x;
  const int w    = tid >> 5, lane = tid & 31;
  const int ln   = lane & 15, hiHalf = lane >> 4;
  const int mb   = w >> 3,  hb = w & 7;
  const int col  = hb * 16 + ln;               // hidden unit owned by this lane

  // zero initial state in LDS
  for (int i = tid; i < BB * 256 / 2; i += 1024)
    ((unsigned int*)hcat)[i] = 0u;

  // ---- TDM: DMA W1cat (global) -> w1lds (LDS), one shot from wave 0 ----
  if (w == 0) {
    typedef __attribute__((address_space(3))) __bf16 lds_bf16;
    unsigned lds_off = (unsigned)(unsigned long long)(lds_bf16*)w1lds;
    unsigned long long ga = (unsigned long long)w1cat;
    // D# group0: count=1 | lds_addr | global_addr(57b) | type=2
    u32x4 g0 = { 1u, lds_off, (unsigned)ga,
                 (unsigned)(ga >> 32) | 0x80000000u };
    // D# group1 (4-byte elements): tensor 128x512 dwords, tile 128x512,
    // dim0 stride 128 dwords. data_size=2 (4B) at bits[17:16].
    i32x8 g1 = { (int)0x00020000u,        // wg_mask=0, data_size=4B
                 (int)(128u << 16),       // tensor_dim0[15:0]=128
                 (int)(512u << 16),       // tensor_dim1[15:0]=512
                 (int)(128u << 16),       // tile_dim0=128
                 (int)512,                // tile_dim1=512, tile_dim2=0
                 (int)128,                // tensor_dim0_stride=128
                 0, 0 };
    i32x4 gz4 = {0, 0, 0, 0};
#if __clang_major__ >= 23
    i32x8 gz8 = {0, 0, 0, 0, 0, 0, 0, 0};
    __builtin_amdgcn_tensor_load_to_lds(g0, g1, gz4, gz4, gz8, 0);
#else
    __builtin_amdgcn_tensor_load_to_lds(g0, g1, gz4, gz4, 0);
#endif
    __builtin_amdgcn_s_wait_tensorcnt(0);
  }

  float bias0g[4], bias1g[4], wi0g[4];
  const __bf16* bp0[4];
  const __bf16* bp1[4];
#pragma unroll
  for (int g = 0; g < 4; ++g) {
    int gc = g * HH + col;
    bias0g[g] = b_ih0[gc] + b_hh0[gc];
    bias1g[g] = b_ih1[gc] + b_hh1[gc];
    wi0g[g]   = w_ih0[gc];
    bp0[g] = whh0 + gc * HH;        // layer-0 B rows: global (L2-resident)
    bp1[g] = w1lds + gc * 256;      // layer-1 B rows: LDS
  }

  float c0r[8], c1r[8];
#pragma unroll
  for (int i = 0; i < 8; ++i) { c0r[i] = 0.f; c1r[i] = 0.f; }

  __syncthreads();   // hcat zeroed + w1lds DMA complete

  const int arow = (mb * 16 + ln) * 256;       // LDS A row base (elements)
  const v8f vzero = {0.f,0.f,0.f,0.f,0.f,0.f,0.f,0.f};

  for (int s = 0; s < NSTEP; ++s) {
    const int f = s & 31, tt = s >> 5;
    if (tid < BB) xbuf[tid] = x_in[tid * (FF * TT) + f * TT + tt];

    // ---- layer 0 GEMM: h0(64x128) @ W_hh0^T(128x512), K=128 ----
    v8f acc0[4] = {vzero, vzero, vzero, vzero};
#pragma unroll
    for (int j = 0; j < 4; ++j) {
      int kA = 32 * j + hiHalf * 8;
      BF16Frag a;
      const uint4v* pa = (const uint4v*)(hcat + arow + kA);
      a.u[0] = pa[0]; a.u[1] = pa[2];
      int kB = 32 * j + hiHalf * 8;
#pragma unroll
      for (int g = 0; g < 4; ++g) {
        BF16Frag b;
        const uint4v* pb = (const uint4v*)(bp0[g] + kB);
        b.u[0] = pb[0]; b.u[1] = pb[2];
        acc0[g] = __builtin_amdgcn_wmma_f32_16x16x32_bf16(false, a.v, false, b.v,
                                                          (short)0, acc0[g], false, false);
      }
    }
    __syncthreads();   // all A reads of h0 done; xbuf visible

    // ---- layer 0 cell (pure VALU, c0 in registers) ----
#pragma unroll
    for (int i = 0; i < 8; ++i) {
      int m = mb * 16 + i + 8 * hiHalf;        // batch row of this element
      float xv = xbuf[m];
      float gi = acc0[0][i] + bias0g[0] + xv * wi0g[0];
      float gf = acc0[1][i] + bias0g[1] + xv * wi0g[1];
      float gz = acc0[2][i] + bias0g[2] + xv * wi0g[2];
      float go = acc0[3][i] + bias0g[3] + xv * wi0g[3];
      float c  = sigf(gf) * c0r[i] + sigf(gi) * tanh_fast(gz);
      c0r[i]   = c;
      float h  = sigf(go) * tanh_fast(c);
      hcat[m * 256 + col] = (__bf16)h;         // new h0
    }
    __syncthreads();   // new h0 visible before layer-1 GEMM

    // ---- layer 1 GEMM: [h0n|h1](64x256) @ [Wih1;Whh1]^T(256x512), K=256 ----
    // B operands come from LDS (staged once by the TDM).
    v8f acc1[4] = {vzero, vzero, vzero, vzero};
#pragma unroll
    for (int j = 0; j < 8; ++j) {
      int kA = 32 * j + hiHalf * 8;
      BF16Frag a;
      const uint4v* pa = (const uint4v*)(hcat + arow + kA);
      a.u[0] = pa[0]; a.u[1] = pa[2];
      int kB = 32 * j + hiHalf * 8;
#pragma unroll
      for (int g = 0; g < 4; ++g) {
        BF16Frag b;
        const uint4v* pb = (const uint4v*)(bp1[g] + kB);
        b.u[0] = pb[0]; b.u[1] = pb[2];
        acc1[g] = __builtin_amdgcn_wmma_f32_16x16x32_bf16(false, a.v, false, b.v,
                                                          (short)0, acc1[g], false, false);
      }
    }
    __syncthreads();   // all reads of h1 done before overwrite

    // ---- layer 1 cell + sampled output ----
#pragma unroll
    for (int i = 0; i < 8; ++i) {
      int m = mb * 16 + i + 8 * hiHalf;
      float gi = acc1[0][i] + bias1g[0];
      float gf = acc1[1][i] + bias1g[1];
      float gz = acc1[2][i] + bias1g[2];
      float go = acc1[3][i] + bias1g[3];
      float c  = sigf(gf) * c1r[i] + sigf(gi) * tanh_fast(gz);
      c1r[i]   = c;
      float h  = sigf(go) * tanh_fast(c);
      hcat[m * 256 + HH + col] = (__bf16)h;    // new h1
      if (f == FF - 1)                          // temp_emb sample every 32 steps
        out_tmp[tt * (BB * HH) + m * HH + col] = h;
    }
    __syncthreads();   // new h1 visible for next step
  }
}

// ---------------------------------------------------------------------------
// Launcher
// ---------------------------------------------------------------------------
extern "C" void kernel_launch(void* const* d_in, const int* in_sizes, int n_in,
                              void* d_out, int out_size, void* d_ws, size_t ws_size,
                              hipStream_t stream) {
  const float* x    = (const float*)d_in[0];   // (B,F,T)
  const float* Ws   = (const float*)d_in[1];   // (F,T,E)
  const float* bs   = (const float*)d_in[2];   // (F,E)
  const float* Wih0 = (const float*)d_in[3];   // (512,1)
  const float* Whh0 = (const float*)d_in[4];   // (512,128)
  const float* bih0 = (const float*)d_in[5];
  const float* bhh0 = (const float*)d_in[6];
  const float* Wih1 = (const float*)d_in[7];   // (512,128)
  const float* Whh1 = (const float*)d_in[8];   // (512,128)
  const float* bih1 = (const float*)d_in[9];
  const float* bhh1 = (const float*)d_in[10];

  // Workspace layout (bf16 staging, ~1.8 MB total)
  __bf16* whh0b = (__bf16*)d_ws;                 // 512*128
  __bf16* w1cat = whh0b + GG * HH;               // 512*256
  __bf16* wstT  = w1cat + GG * 256;              // 32*112*128
  __bf16* xbf   = wstT + FF * EPAD * TT;         // 64*32*128

  float* out_sp  = (float*)d_out;                // (B,F,E)
  float* out_tmp = out_sp + BB * FF * EE;        // (T,B,H)

  k_cvt_bf16  <<<(GG * HH) / 256,        256, 0, stream>>>(Whh0, whh0b, GG * HH);
  k_prep_w1cat<<<(GG * 256) / 256,       256, 0, stream>>>(Wih1, Whh1, w1cat);
  k_prep_wst  <<<(FF * EPAD * TT) / 256, 256, 0, stream>>>(Ws, wstT);
  k_cvt_bf16  <<<(BB * FF * TT) / 256,   256, 0, stream>>>(x, xbf, BB * FF * TT);

  k_spatial<<<112, 256, 0, stream>>>(xbf, wstT, bs, out_sp);

  // Allow 256 KB of dynamic LDS for the persistent LSTM workgroup
  (void)hipFuncSetAttribute((const void*)k_lstm,
                            hipFuncAttributeMaxDynamicSharedMemorySize,
                            W1LDS_BYTES);
  k_lstm<<<1, 1024, W1LDS_BYTES, stream>>>(x, whh0b, w1cat, Wih0,
                                           bih0, bhh0, bih1, bhh1, out_tmp);
}
